// Attention_833223655514
// MI455X (gfx1250) — compile-verified, matrix operand order checked by reading, not run on hardware
//
#include <hip/hip_runtime.h>

#define B_  2
#define T_  2048
#define D_  2048
#define H_  16
#define HD_ 128

typedef __bf16 bf16x16 __attribute__((ext_vector_type(16)));
typedef float  f32x8   __attribute__((ext_vector_type(8)));
typedef unsigned int u32x4 __attribute__((ext_vector_type(4)));
typedef int          i32x4 __attribute__((ext_vector_type(4)));
typedef int          i32x8 __attribute__((ext_vector_type(8)));

#ifndef __has_builtin
#define __has_builtin(x) 0
#endif
#if __has_builtin(__builtin_amdgcn_tensor_load_to_lds) && \
    __has_builtin(__builtin_amdgcn_s_wait_tensorcnt)
#define HAVE_TDM 1
#else
#define HAVE_TDM 0
#endif

struct __attribute__((aligned(16))) U4 { unsigned int x, y, z, w; };
struct U4x2 { U4 lo, hi; };

__device__ __forceinline__ unsigned short f2bf(float f) {
    unsigned int u = __float_as_uint(f);
    u += 0x7FFFu + ((u >> 16) & 1u);          // round-to-nearest-even
    return (unsigned short)(u >> 16);
}
__device__ __forceinline__ float bf2f(unsigned short h) {
    return __uint_as_float(((unsigned int)h) << 16);
}

__device__ __forceinline__ bf16x16 load_frag(const unsigned short* p0,
                                             const unsigned short* p1) {
    U4x2 t;
    t.lo = *(const U4*)p0;
    t.hi = *(const U4*)p1;
    return __builtin_bit_cast(bf16x16, t);
}

__device__ __forceinline__ f32x8 zero8() {
    f32x8 z = {0.f, 0.f, 0.f, 0.f, 0.f, 0.f, 0.f, 0.f};
    return z;
}

#if HAVE_TDM
// LDS byte offset of a __shared__ object: generic LDS address keeps the
// wave-relative LDS offset in its low 32 bits (ISA aperture mapping).
__device__ __forceinline__ unsigned lds_off(const void* p) {
    return (unsigned)(uintptr_t)p;
}

// Issue a TDM 2-D tile load: tile_w x tile_h bf16 elements, row stride
// `stride_elems`, from global `gptr` into LDS at byte offset `lds_addr`
// (packed row-major, tile_w contiguous elements per row).
__device__ __forceinline__ void tdm_load_2d(unsigned lds_addr,
                                            const unsigned short* gptr,
                                            unsigned tile_w, unsigned tile_h,
                                            unsigned stride_elems) {
    unsigned long long ga = (unsigned long long)(uintptr_t)gptr;
    // ---- D# group 0 (ISA 8.3): count=1, lds_addr, 57-bit global addr, type=2
    u32x4 g0;
    g0[0] = 1u;                                   // count = 1 (valid descriptor)
    g0[1] = lds_addr;                             // LDS byte address
    g0[2] = (unsigned)(ga & 0xFFFFFFFFu);         // global_addr[31:0]
    g0[3] = (unsigned)((ga >> 32) & 0x01FFFFFFu)  // global_addr[56:32]
          | (2u << 30);                           // type = 2 ("image")
    // ---- D# group 1 (ISA 8.4)
    const unsigned td0 = stride_elems;            // tensor_dim0 = true row length
    const unsigned td1 = 0x000FFFFFu;             // tensor_dim1: generous (in-bounds tiles)
    i32x8 g1;
    g1[0] = (int)(1u << 16);                                      // data_size=2B
    g1[1] = (int)((td0 & 0xFFFFu) << 16);                         // barrier_addr=0 | td0 lo
    g1[2] = (int)((td0 >> 16) | ((td1 & 0xFFFFu) << 16));         // td0 hi | td1 lo
    g1[3] = (int)((td1 >> 16) | (tile_w << 16));                  // td1 hi | tile_dim0
    g1[4] = (int)(tile_h & 0xFFFFu);                              // tile_dim1 | tile_dim2=0
    g1[5] = (int)stride_elems;                                    // dim0_stride[31:0]
    g1[6] = 0;                                                    // stride hi | dim1_stride
    g1[7] = 0;
    i32x4 z4 = {0, 0, 0, 0};
#if __clang_major__ >= 23
    i32x8 z8 = {0, 0, 0, 0, 0, 0, 0, 0};
    __builtin_amdgcn_tensor_load_to_lds(g0, g1, z4, z4, z8, 0);
#else
    __builtin_amdgcn_tensor_load_to_lds(g0, g1, z4, z4, 0);
#endif
}
#endif  // HAVE_TDM

// ---------------------------------------------------------------- cast f32->bf16
__global__ __launch_bounds__(256) void cast_f32_bf16(const float* __restrict__ in,
                                                     unsigned short* __restrict__ out,
                                                     size_t n) {
    size_t i = (size_t)blockIdx.x * blockDim.x + threadIdx.x;
    size_t stride = (size_t)gridDim.x * blockDim.x;
    for (; i < n; i += stride) out[i] = f2bf(in[i]);
}

// ---------------------------------------------------------------- GEMM: C = A * W^T
// A: [M,K] bf16 row-major, W: [N,K] bf16 row-major (so both K-contiguous).
// Block = 256 thr = 8 waves; macrotile 128x128; wave tile 32(M) x 64(N).
// TDM path: wave 0 DMAs the next 128x32 panels into the alternate LDS buffer
// while all waves compute on the current one (double-buffered, TENSORcnt-paced).
template <bool OUT_F32>
__global__ __launch_bounds__(256) void gemm_bt(
    const unsigned short* __restrict__ A,
    const unsigned short* __restrict__ W,
    void* __restrict__ C,
    int M, int N, int K) {
#if HAVE_TDM
    __shared__ __attribute__((aligned(16))) unsigned short As[2][128 * 32];
    __shared__ __attribute__((aligned(16))) unsigned short Ws[2][128 * 32];
#else
    __shared__ __attribute__((aligned(16))) unsigned short As[1][128 * 32];
    __shared__ __attribute__((aligned(16))) unsigned short Ws[1][128 * 32];
#endif
    const int tid  = threadIdx.x;
    const int lane = tid & 31;
    const int w    = tid >> 5;
    const int wm   = w & 3;              // 0..3 -> M sub-tile
    const int wn   = w >> 2;             // 0..1 -> N sub-tile
    const int ln   = lane & 15;
    const int lo   = (lane < 16) ? 0 : 8;    // K chunk offset per lane half
    const int rofs = (lane < 16) ? 0 : 8;    // C row offset per lane half
    const int gm0  = blockIdx.y * 128;
    const int gn0  = blockIdx.x * 128;

    f32x8 acc[2][4];
#pragma unroll
    for (int mi = 0; mi < 2; ++mi)
#pragma unroll
        for (int ni = 0; ni < 4; ++ni) acc[mi][ni] = zero8();

    const int nk = K / 32;
#if HAVE_TDM
    if (w == 0) {
        tdm_load_2d(lds_off(As[0]), A + (size_t)gm0 * K, 32, 128, (unsigned)K);
        tdm_load_2d(lds_off(Ws[0]), W + (size_t)gn0 * K, 32, 128, (unsigned)K);
    }
#endif
    for (int kt = 0; kt < nk; ++kt) {
        const int kk = kt * 32;
#if HAVE_TDM
        const int cur = kt & 1;
        if (w == 0) {
            if (kt + 1 < nk) {
                const int kn = kk + 32;
                tdm_load_2d(lds_off(As[cur ^ 1]), A + (size_t)gm0 * K + kn,
                            32, 128, (unsigned)K);
                tdm_load_2d(lds_off(Ws[cur ^ 1]), W + (size_t)gn0 * K + kn,
                            32, 128, (unsigned)K);
                __builtin_amdgcn_s_wait_tensorcnt(2);  // current panels landed
            } else {
                __builtin_amdgcn_s_wait_tensorcnt(0);
            }
        }
        __syncthreads();                // panels visible to all waves
        const unsigned short* Asb = As[cur];
        const unsigned short* Wsb = Ws[cur];
#else
        const int cur = 0;
        __syncthreads();
        {
            const int srow = tid >> 1;
            const int sseg = (tid & 1) * 16;
            const unsigned short* ag = A + (size_t)(gm0 + srow) * K + kk + sseg;
            const unsigned short* wg = W + (size_t)(gn0 + srow) * K + kk + sseg;
            *(U4*)&As[cur][srow * 32 + sseg]     = *(const U4*)ag;
            *(U4*)&As[cur][srow * 32 + sseg + 8] = *(const U4*)(ag + 8);
            *(U4*)&Ws[cur][srow * 32 + sseg]     = *(const U4*)wg;
            *(U4*)&Ws[cur][srow * 32 + sseg + 8] = *(const U4*)(wg + 8);
            __builtin_prefetch(ag + 32, 0, 1);
            __builtin_prefetch(wg + 32, 0, 1);
        }
        __syncthreads();
        const unsigned short* Asb = As[cur];
        const unsigned short* Wsb = Ws[cur];
#endif
        bf16x16 af[2];
#pragma unroll
        for (int mi = 0; mi < 2; ++mi) {
            int r = wm * 32 + mi * 16 + ln;
            af[mi] = load_frag(&Asb[r * 32 + lo], &Asb[r * 32 + 16 + lo]);
        }
#pragma unroll
        for (int ni = 0; ni < 4; ++ni) {
            int r = wn * 64 + ni * 16 + ln;
            bf16x16 bfrag = load_frag(&Wsb[r * 32 + lo], &Wsb[r * 32 + 16 + lo]);
#pragma unroll
            for (int mi = 0; mi < 2; ++mi)
                acc[mi][ni] = __builtin_amdgcn_wmma_f32_16x16x32_bf16(
                    false, af[mi], false, bfrag, (short)0, acc[mi][ni], false, false);
        }
        __syncthreads();                // done reading: panel buffer reusable
    }

#pragma unroll
    for (int mi = 0; mi < 2; ++mi) {
#pragma unroll
        for (int ni = 0; ni < 4; ++ni) {
            int gcol = gn0 + wn * 64 + ni * 16 + ln;
#pragma unroll
            for (int r = 0; r < 8; ++r) {
                int grow = gm0 + wm * 32 + mi * 16 + r + rofs;
                if (OUT_F32)
                    ((float*)C)[(size_t)grow * N + gcol] = acc[mi][ni][r];
                else
                    ((unsigned short*)C)[(size_t)grow * N + gcol] = f2bf(acc[mi][ni][r]);
            }
        }
    }
}

// ---------------------------------------------------------------- RoPE (bf16 in-place)
__global__ __launch_bounds__(256) void rope_k(unsigned short* __restrict__ qk,
                                              const float* __restrict__ fc) {
    size_t idx = (size_t)blockIdx.x * blockDim.x + threadIdx.x;  // B*T*H*64 total
    int i = (int)(idx & 63);
    int h = (int)((idx >> 6) & 15);
    int t = (int)((idx >> 10) & 2047);
    int b = (int)(idx >> 21);
    if (b >= B_) return;
    size_t base = ((size_t)(b * T_ + t)) * D_ + (size_t)h * HD_ + 2 * i;
    float c = fc[((size_t)t * 64 + i) * 2];
    float s = fc[((size_t)t * 64 + i) * 2 + 1];
    float x0 = bf2f(qk[base]);
    float x1 = bf2f(qk[base + 1]);
    qk[base]     = f2bf(x0 * c - x1 * s);
    qk[base + 1] = f2bf(x0 * s + x1 * c);
}

// ---------------------------------------------------------------- flash attention
// Block = 256 thr = 8 waves, 128 queries/block, one (b,h) per block.
// Each wave owns 16 full query rows -> online softmax entirely within a wave.
// TDM path: wave 0 DMAs the 32x128 K panel while all threads stage V^T manually.
__global__ __launch_bounds__(256) void flash_attn(
    const unsigned short* __restrict__ Q,
    const unsigned short* __restrict__ Km,
    const unsigned short* __restrict__ Vm,
    unsigned short* __restrict__ O) {
    __shared__ __attribute__((aligned(16))) unsigned short Ks[32 * 128];   // keys x hd
    __shared__ __attribute__((aligned(16))) unsigned short Vt[128 * 32];   // hd x keys
    __shared__ __attribute__((aligned(16))) unsigned short Ps[8 * 16 * 32];// per-wave P
    const int tid  = threadIdx.x;
    const int lane = tid & 31;
    const int w    = tid >> 5;
    const int ln   = lane & 15;
    const int lo   = (lane < 16) ? 0 : 8;
    const int rofs = (lane < 16) ? 0 : 8;
    const int h = blockIdx.y, b = blockIdx.z;
    const size_t base = (size_t)b * T_ * D_ + (size_t)h * HD_;
    const int q0 = blockIdx.x * 128 + w * 16;

    bf16x16 qa[4];
    {
        const unsigned short* qrow = Q + base + (size_t)(q0 + ln) * D_;
#pragma unroll
        for (int ks = 0; ks < 4; ++ks)
            qa[ks] = load_frag(qrow + ks * 32 + lo, qrow + ks * 32 + 16 + lo);
    }

    f32x8 o[8];
#pragma unroll
    for (int t = 0; t < 8; ++t) o[t] = zero8();
    float mrun[8], lrun[8], alpha[8];
#pragma unroll
    for (int r = 0; r < 8; ++r) { mrun[r] = -3.0e38f; lrun[r] = 0.f; }
    const float scale = 0.08838834764831845f;  // 1/sqrt(128)
    unsigned short* Pw = Ps + w * (16 * 32);

    for (int j = 0; j < T_ / 32; ++j) {
        const int kb0 = j * 32;
        __syncthreads();
#if HAVE_TDM
        if (w == 0)  // async DMA of K panel overlaps manual V^T staging below
            tdm_load_2d(lds_off(Ks), Km + base + (size_t)kb0 * D_, 128, 32, D_);
#endif
        {
            const int row = tid >> 3;
            const int seg = (tid & 7) * 16;
#if !HAVE_TDM
            const unsigned short* ksrc = Km + base + (size_t)(kb0 + row) * D_ + seg;
            *(U4*)&Ks[row * 128 + seg]     = *(const U4*)ksrc;
            *(U4*)&Ks[row * 128 + seg + 8] = *(const U4*)(ksrc + 8);
#endif
            const unsigned short* vsrc = Vm + base + (size_t)(kb0 + row) * D_ + seg;
            union { U4 u[2]; unsigned short s[16]; } vv;
            vv.u[0] = *(const U4*)vsrc;
            vv.u[1] = *(const U4*)(vsrc + 8);
#pragma unroll
            for (int i = 0; i < 16; ++i)
                Vt[(seg + i) * 32 + row] = vv.s[i];
        }
#if HAVE_TDM
        if (w == 0) __builtin_amdgcn_s_wait_tensorcnt(0);
#endif
        __syncthreads();

        // S = Q K^T * scale  (two 16x16 tiles over 32 keys)
        f32x8 s[2];
#pragma unroll
        for (int nt = 0; nt < 2; ++nt) {
            s[nt] = zero8();
#pragma unroll
            for (int ks = 0; ks < 4; ++ks) {
                const int key = nt * 16 + ln;
                bf16x16 kf = load_frag(&Ks[key * 128 + ks * 32 + lo],
                                       &Ks[key * 128 + ks * 32 + 16 + lo]);
                s[nt] = __builtin_amdgcn_wmma_f32_16x16x32_bf16(
                    false, qa[ks], false, kf, (short)0, s[nt], false, false);
            }
            s[nt] *= scale;
        }

        // online softmax: rows live in 16-lane halves -> xor-shuffle reductions
#pragma unroll
        for (int r = 0; r < 8; ++r) {
            float mx = fmaxf(s[0][r], s[1][r]);
            mx = fmaxf(mx, __shfl_xor(mx, 8, 32));
            mx = fmaxf(mx, __shfl_xor(mx, 4, 32));
            mx = fmaxf(mx, __shfl_xor(mx, 2, 32));
            mx = fmaxf(mx, __shfl_xor(mx, 1, 32));
            float mnew = fmaxf(mrun[r], mx);
            alpha[r] = __expf(mrun[r] - mnew);
            mrun[r] = mnew;
            float p0 = __expf(s[0][r] - mnew);
            float p1 = __expf(s[1][r] - mnew);
            s[0][r] = p0; s[1][r] = p1;
            float rs = p0 + p1;
            rs += __shfl_xor(rs, 8, 32);
            rs += __shfl_xor(rs, 4, 32);
            rs += __shfl_xor(rs, 2, 32);
            rs += __shfl_xor(rs, 1, 32);
            lrun[r] = lrun[r] * alpha[r] + rs;
        }
#pragma unroll
        for (int t = 0; t < 8; ++t)
#pragma unroll
            for (int r = 0; r < 8; ++r) o[t][r] *= alpha[r];

        // C-layout P -> row-major LDS -> reload as A fragment
#pragma unroll
        for (int nt = 0; nt < 2; ++nt)
#pragma unroll
            for (int r = 0; r < 8; ++r)
                Pw[(r + rofs) * 32 + nt * 16 + ln] = f2bf(s[nt][r]);
        bf16x16 pa = load_frag(&Pw[ln * 32 + lo], &Pw[ln * 32 + 16 + lo]);

        // O += P * V  (8 output tiles over hd=128, K-dim = 32 keys)
#pragma unroll
        for (int t = 0; t < 8; ++t) {
            const int d = t * 16 + ln;
            bf16x16 vf = load_frag(&Vt[d * 32 + lo], &Vt[d * 32 + 16 + lo]);
            o[t] = __builtin_amdgcn_wmma_f32_16x16x32_bf16(
                false, pa, false, vf, (short)0, o[t], false, false);
        }
    }

#pragma unroll
    for (int r = 0; r < 8; ++r) lrun[r] = 1.0f / lrun[r];
#pragma unroll
    for (int t = 0; t < 8; ++t) {
        const int d = t * 16 + ln;
#pragma unroll
        for (int r = 0; r < 8; ++r) {
            const int qrow = q0 + r + rofs;
            O[base + (size_t)qrow * D_ + d] = f2bf(o[t][r] * lrun[r]);
        }
    }
}

// ---------------------------------------------------------------- launch
extern "C" void kernel_launch(void* const* d_in, const int* in_sizes, int n_in,
                              void* d_out, int out_size, void* d_ws, size_t ws_size,
                              hipStream_t stream) {
    const float* x  = (const float*)d_in[0];
    const float* fc = (const float*)d_in[1];
    const float* wq = (const float*)d_in[2];
    const float* wk = (const float*)d_in[3];
    const float* wv = (const float*)d_in[4];
    const float* wo = (const float*)d_in[5];
    float* out = (float*)d_out;

    const size_t MD = (size_t)B_ * T_ * D_;  // 8388608
    const size_t DD = (size_t)D_ * D_;       // 4194304
    unsigned short* ws  = (unsigned short*)d_ws;
    unsigned short* xb  = ws;
    unsigned short* wqb = xb + MD;
    unsigned short* wkb = wqb + DD;
    unsigned short* wvb = wkb + DD;
    unsigned short* wob = wvb + DD;
    unsigned short* qb  = wob + DD;
    unsigned short* kb  = qb + MD;
    unsigned short* vb  = kb + MD;
    unsigned short* ab  = vb + MD;

    dim3 blk(256);
    cast_f32_bf16<<<4096, blk, 0, stream>>>(x, xb, MD);
    cast_f32_bf16<<<2048, blk, 0, stream>>>(wq, wqb, DD);
    cast_f32_bf16<<<2048, blk, 0, stream>>>(wk, wkb, DD);
    cast_f32_bf16<<<2048, blk, 0, stream>>>(wv, wvb, DD);
    cast_f32_bf16<<<2048, blk, 0, stream>>>(wo, wob, DD);

    dim3 ggrid(D_ / 128, (B_ * T_) / 128);   // (16, 32)
    gemm_bt<false><<<ggrid, blk, 0, stream>>>(xb, wqb, qb, B_ * T_, D_, D_);
    gemm_bt<false><<<ggrid, blk, 0, stream>>>(xb, wkb, kb, B_ * T_, D_, D_);
    gemm_bt<false><<<ggrid, blk, 0, stream>>>(xb, wvb, vb, B_ * T_, D_, D_);

    const size_t nrope = (size_t)B_ * T_ * H_ * (HD_ / 2);   // 2^22
    rope_k<<<(unsigned)(nrope / 256), blk, 0, stream>>>(qb, fc);
    rope_k<<<(unsigned)(nrope / 256), blk, 0, stream>>>(kb, fc);

    dim3 fgrid(T_ / 128, H_, B_);            // (16, 16, 2)
    flash_attn<<<fgrid, blk, 0, stream>>>(qb, kb, vb, ab);

    gemm_bt<true><<<ggrid, blk, 0, stream>>>(ab, wob, out, B_ * T_, D_, D_);
}
